// AttentionLSTMCell_19731079758068
// MI455X (gfx1250) — compile-verified, matrix operand order checked by reading, not run on hardware
//
#include <hip/hip_runtime.h>
#include <hip/hip_bf16.h>

#define B_ 32
#define T_ 128
#define D_ 512
#define U_ 512
#define G_ 2048   // 4*U

typedef __attribute__((ext_vector_type(16))) __bf16 v16bf;
typedef __attribute__((ext_vector_type(8)))  float  v8f;

union FragBF {
    v16bf v;
    unsigned short s[16];
    uint4 q[2];
};

__device__ __forceinline__ unsigned short f2bf(float f) {
    unsigned int u = __float_as_uint(f);
    unsigned int r = (u + 0x7FFFu + ((u >> 16) & 1u)) >> 16;   // round-to-nearest-even
    return (unsigned short)r;
}

// K index of fragment element e (even e; e and e+1 are consecutive K) for a lane half.
__device__ __forceinline__ int kmap(int e, int half) {
    return (e & 7) + (half << 3) + ((e & 8) << 1);
}

__device__ __forceinline__ float hsig(float x) {
    return fminf(fmaxf(0.2f * x + 0.5f, 0.0f), 1.0f);
}

// ---------------- packing kernels (run once) ----------------

// Pack f32 (M x K) row-major LHS into WMMA A-fragment tiles of bf16.
// Layout: tile (mt,kt) -> Ap[((mt*KT+kt)*32 + lane)*16 + e]
__global__ void pack_a_bf16(const float* __restrict__ A, unsigned short* __restrict__ Ap,
                            int M, int K) {
    int gid = blockIdx.x * blockDim.x + threadIdx.x;
    int KT = K >> 5;
    int total = (M >> 4) * KT * 32;
    if (gid >= total) return;
    int lane = gid & 31;
    int tile = gid >> 5;          // = mt*KT + kt
    int kt   = tile % KT;
    int half = lane >> 4;
    int m    = ((tile / KT) << 4) + (lane & 15);
    const float* row = A + (size_t)m * K + (kt << 5);
    FragBF f;
#pragma unroll
    for (int j = 0; j < 8; ++j) {
        int kb = kmap(2 * j, half);
        float2 v = *reinterpret_cast<const float2*>(row + kb);
        f.s[2 * j]     = f2bf(v.x);
        f.s[2 * j + 1] = f2bf(v.y);
    }
    uint4* dst = reinterpret_cast<uint4*>(Ap + ((size_t)tile * 32 + lane) * 16);
    dst[0] = f.q[0];
    dst[1] = f.q[1];
}

// Pack f32 (K x N) row-major weight into WMMA B-fragment tiles of bf16.
// Layout: tile (kt,nt) -> Wp[((kt*NT+nt)*32 + lane)*16 + e]
__global__ void pack_b_bf16(const float* __restrict__ W, unsigned short* __restrict__ Wp,
                            int K, int N) {
    int gid = blockIdx.x * blockDim.x + threadIdx.x;
    int NT = N >> 4;
    int total = (K >> 5) * NT * 32;
    if (gid >= total) return;
    int lane = gid & 31;
    int tile = gid >> 5;          // = kt*NT + nt
    int nt   = tile % NT;
    int kt   = tile / NT;
    int half = lane >> 4;
    int n    = (nt << 4) + (lane & 15);
    FragBF f;
#pragma unroll
    for (int j = 0; j < 8; ++j) {
        int kb = (kt << 5) + kmap(2 * j, half);
        f.s[2 * j]     = f2bf(W[(size_t)kb * N + n]);
        f.s[2 * j + 1] = f2bf(W[(size_t)(kb + 1) * N + n]);
    }
    uint4* dst = reinterpret_cast<uint4*>(Wp + ((size_t)tile * 32 + lane) * 16);
    dst[0] = f.q[0];
    dst[1] = f.q[1];
}

// ---------------- GEMM kernels ----------------

// C(M,N) = Apacked @ Bpacked (+ bias row vector).  One wave per 16x16 tile.
__global__ void gemm_bf16_packed(const unsigned short* __restrict__ Ap,
                                 const unsigned short* __restrict__ Bp,
                                 const float* __restrict__ bias,
                                 float* __restrict__ C,
                                 int M, int N, int K) {
    int wave = blockIdx.x * (blockDim.x >> 5) + (threadIdx.x >> 5);
    int lane = threadIdx.x & 31;
    int MT = M >> 4, NT = N >> 4, KT = K >> 5;
    if (wave >= MT * NT) return;
    int mt = wave / NT, nt = wave % NT;
    v8f acc = {};
    for (int kt = 0; kt < KT; ++kt) {
        FragBF a, b;
        const uint4* pa = reinterpret_cast<const uint4*>(
            Ap + (((size_t)(mt * KT + kt) * 32) + lane) * 16);
        a.q[0] = pa[0]; a.q[1] = pa[1];
        const uint4* pb = reinterpret_cast<const uint4*>(
            Bp + (((size_t)(kt * NT + nt) * 32) + lane) * 16);
        b.q[0] = pb[0]; b.q[1] = pb[1];
        acc = __builtin_amdgcn_wmma_f32_16x16x32_bf16(false, a.v, false, b.v,
                                                      (short)0, acc, false, false);
    }
    int half = lane >> 4;
    int n = (nt << 4) + (lane & 15);
    float bv = bias ? bias[n] : 0.0f;
#pragma unroll
    for (int r = 0; r < 8; ++r) {
        int m = (mt << 4) + r + (half << 3);
        C[(size_t)m * N + n] = acc[r] + bv;
    }
}

// C(M,N) = A(f32, row-major, converted inline) @ Bpacked.  For hU = h @ attention_U.
__global__ void gemm_af32_bf16(const float* __restrict__ A,
                               const unsigned short* __restrict__ Bp,
                               float* __restrict__ C,
                               int M, int N, int K) {
    int wave = blockIdx.x * (blockDim.x >> 5) + (threadIdx.x >> 5);
    int lane = threadIdx.x & 31;
    int MT = M >> 4, NT = N >> 4, KT = K >> 5;
    if (wave >= MT * NT) return;
    int mt = wave / NT, nt = wave % NT;
    int half = lane >> 4;
    int mrow = (mt << 4) + (lane & 15);
    v8f acc = {};
    for (int kt = 0; kt < KT; ++kt) {
        FragBF a, b;
        const float* row = A + (size_t)mrow * K + (kt << 5);
#pragma unroll
        for (int j = 0; j < 8; ++j) {
            int kb = kmap(2 * j, half);
            float2 v = *reinterpret_cast<const float2*>(row + kb);
            a.s[2 * j]     = f2bf(v.x);
            a.s[2 * j + 1] = f2bf(v.y);
        }
        const uint4* pb = reinterpret_cast<const uint4*>(
            Bp + (((size_t)(kt * NT + nt) * 32) + lane) * 16);
        b.q[0] = pb[0]; b.q[1] = pb[1];
        acc = __builtin_amdgcn_wmma_f32_16x16x32_bf16(false, a.v, false, b.v,
                                                      (short)0, acc, false, false);
    }
    int n = (nt << 4) + (lane & 15);
#pragma unroll
    for (int r = 0; r < 8; ++r) {
        int m = (mt << 4) + r + (half << 3);
        C[(size_t)m * N + n] = acc[r];
    }
}

// gates(32,2048) = xk[:,t,:] + h @ Rpacked + z @ Apacked
__global__ void gates_gemm(const float* __restrict__ h, const float* __restrict__ z,
                           const unsigned short* __restrict__ Rp,
                           const unsigned short* __restrict__ Ap,
                           const float* __restrict__ xk,
                           float* __restrict__ gates, int t) {
    const int NT = G_ >> 4;   // 128
    const int KT = U_ >> 5;   // 16
    int wave = blockIdx.x * (blockDim.x >> 5) + (threadIdx.x >> 5);
    int lane = threadIdx.x & 31;
    if (wave >= (B_ >> 4) * NT) return;
    int mt = wave / NT, nt = wave % NT;
    int half = lane >> 4;
    int mrow = (mt << 4) + (lane & 15);
    v8f acc = {};
    for (int kt = 0; kt < KT; ++kt) {          // h @ R
        FragBF a, b;
        const float* row = h + (size_t)mrow * U_ + (kt << 5);
#pragma unroll
        for (int j = 0; j < 8; ++j) {
            int kb = kmap(2 * j, half);
            float2 v = *reinterpret_cast<const float2*>(row + kb);
            a.s[2 * j] = f2bf(v.x); a.s[2 * j + 1] = f2bf(v.y);
        }
        const uint4* pb = reinterpret_cast<const uint4*>(
            Rp + (((size_t)(kt * NT + nt) * 32) + lane) * 16);
        b.q[0] = pb[0]; b.q[1] = pb[1];
        acc = __builtin_amdgcn_wmma_f32_16x16x32_bf16(false, a.v, false, b.v,
                                                      (short)0, acc, false, false);
    }
    for (int kt = 0; kt < KT; ++kt) {          // z @ A
        FragBF a, b;
        const float* row = z + (size_t)mrow * D_ + (kt << 5);
#pragma unroll
        for (int j = 0; j < 8; ++j) {
            int kb = kmap(2 * j, half);
            float2 v = *reinterpret_cast<const float2*>(row + kb);
            a.s[2 * j] = f2bf(v.x); a.s[2 * j + 1] = f2bf(v.y);
        }
        const uint4* pb = reinterpret_cast<const uint4*>(
            Ap + (((size_t)(kt * NT + nt) * 32) + lane) * 16);
        b.q[0] = pb[0]; b.q[1] = pb[1];
        acc = __builtin_amdgcn_wmma_f32_16x16x32_bf16(false, a.v, false, b.v,
                                                      (short)0, acc, false, false);
    }
    int n = (nt << 4) + (lane & 15);
#pragma unroll
    for (int r = 0; r < 8; ++r) {
        int m = (mt << 4) + r + (half << 3);            // batch row b
        gates[(size_t)m * G_ + n] = acc[r] + xk[((size_t)m * T_ + t) * G_ + n];
    }
}

// ---------------- scan-step elementwise kernels ----------------

// scores[b,t'] = sum_u tanh(att_x[b,t',u] + hU[b,u]) * V[u].  One wave per (b,t').
__global__ void scores_kernel(const float* __restrict__ attx, const float* __restrict__ hU,
                              const float* __restrict__ V, float* __restrict__ scores) {
    int wave = blockIdx.x * (blockDim.x >> 5) + (threadIdx.x >> 5);
    int lane = threadIdx.x & 31;
    if (wave >= B_ * T_) return;
    int b = wave / T_;
    const float* ax = attx + (size_t)wave * U_;
    const float* hb = hU + (size_t)b * U_;
    float partial = 0.0f;
#pragma unroll 4
    for (int i = 0; i < U_ / 32; ++i) {
        int u = lane + 32 * i;
        partial += tanhf(ax[u] + hb[u]) * V[u];
    }
#pragma unroll
    for (int off = 16; off > 0; off >>= 1) partial += __shfl_xor(partial, off, 32);
    if (lane == 0) scores[wave] = partial;
}

// alpha = softmax(scores[b,:]); z[b,d] = sum_t alpha[t]*x[b,t,d].  One block per b.
__global__ void softmax_z_kernel(const float* __restrict__ scores,
                                 const float* __restrict__ x, float* __restrict__ z) {
    __shared__ float sh[T_];
    __shared__ float al[T_];
    int b = blockIdx.x, tid = threadIdx.x;                   // blockDim = 128
    float s = scores[b * T_ + tid];
    sh[tid] = s; __syncthreads();
    for (int o = 64; o > 0; o >>= 1) { if (tid < o) sh[tid] = fmaxf(sh[tid], sh[tid + o]); __syncthreads(); }
    float mx = sh[0]; __syncthreads();
    float p = expf(s - mx);
    sh[tid] = p; __syncthreads();
    for (int o = 64; o > 0; o >>= 1) { if (tid < o) sh[tid] += sh[tid + o]; __syncthreads(); }
    float inv = 1.0f / sh[0];
    al[tid] = p * inv;
    __syncthreads();
    const float* xb = x + (size_t)b * T_ * D_;
    for (int d = tid; d < D_; d += 128) {
        float acc = 0.0f;
        for (int tt = 0; tt < T_; ++tt) acc += al[tt] * xb[(size_t)tt * D_ + d];
        z[(size_t)b * D_ + d] = acc;
    }
}

__global__ void lstm_update(const float* __restrict__ gates, float* __restrict__ h,
                            float* __restrict__ c, float* __restrict__ out, int t) {
    int idx = blockIdx.x * blockDim.x + threadIdx.x;
    if (idx >= B_ * U_) return;
    int b = idx >> 9;
    int u = idx & 511;
    const float* g = gates + (size_t)b * G_;
    float i = hsig(g[u]);
    float f = hsig(g[U_ + u]);
    float o = hsig(g[3 * U_ + u]);
    float cn = f * c[idx] + i * tanhf(g[2 * U_ + u]);
    float hn = o * tanhf(cn);
    c[idx] = cn;
    h[idx] = hn;
    out[((size_t)b * T_ + t) * U_ + u] = hn;
}

__global__ void zero_kernel(float* __restrict__ p, int n) {
    int i = blockIdx.x * blockDim.x + threadIdx.x;
    if (i < n) p[i] = 0.0f;
}

// ---------------- launcher ----------------

extern "C" void kernel_launch(void* const* d_in, const int* in_sizes, int n_in,
                              void* d_out, int out_size, void* d_ws, size_t ws_size,
                              hipStream_t stream) {
    (void)in_sizes; (void)n_in; (void)out_size; (void)ws_size;
    const float* x    = (const float*)d_in[0];   // (B,T,D)
    const float* Wk   = (const float*)d_in[1];   // kernel (D,4U)
    const float* Wr   = (const float*)d_in[2];   // recurrent_kernel (U,4U)
    const float* Wa   = (const float*)d_in[3];   // attention_kernel (D,4U)
    const float* attW = (const float*)d_in[4];   // (D,U)
    const float* attU = (const float*)d_in[5];   // (U,U)
    const float* attV = (const float*)d_in[6];   // (U,1)
    const float* bias = (const float*)d_in[7];   // (4U)
    const float* attb = (const float*)d_in[8];   // (U)
    float* out = (float*)d_out;                  // (B,T,U)

    char* ws = (char*)d_ws;
    size_t off = 0;
    auto allocF = [&](size_t n) -> float* {
        float* p = (float*)(ws + off); off += n * sizeof(float); return p; };
    auto allocH = [&](size_t n) -> unsigned short* {
        unsigned short* p = (unsigned short*)(ws + off);
        off += n * sizeof(unsigned short); off = (off + 31) & ~(size_t)31; return p; };

    float* attx   = allocF((size_t)B_ * T_ * U_);   // 2.1M f32
    float* xk     = allocF((size_t)B_ * T_ * G_);   // 8.4M f32
    float* h      = allocF((size_t)B_ * U_);
    float* c      = allocF((size_t)B_ * U_);
    float* hU     = allocF((size_t)B_ * U_);
    float* z      = allocF((size_t)B_ * D_);
    float* sc     = allocF((size_t)B_ * T_);
    float* gates  = allocF((size_t)B_ * G_);
    unsigned short* xpack  = allocH((size_t)B_ * T_ * D_);
    unsigned short* p_kern = allocH((size_t)D_ * G_);
    unsigned short* p_attW = allocH((size_t)D_ * U_);
    unsigned short* p_attU = allocH((size_t)U_ * U_);
    unsigned short* p_R    = allocH((size_t)U_ * G_);
    unsigned short* p_A    = allocH((size_t)D_ * G_);

    // ---- one-time packing into WMMA fragment layouts ----
    pack_a_bf16<<<512, 256, 0, stream>>>(x, xpack, B_ * T_, D_);            // 131072 thr
    pack_b_bf16<<<256, 256, 0, stream>>>(Wk,   p_kern, D_, G_);
    pack_b_bf16<<< 64, 256, 0, stream>>>(attW, p_attW, D_, U_);
    pack_b_bf16<<< 64, 256, 0, stream>>>(attU, p_attU, U_, U_);
    pack_b_bf16<<<256, 256, 0, stream>>>(Wr,   p_R,    U_, G_);
    pack_b_bf16<<<256, 256, 0, stream>>>(Wa,   p_A,    D_, G_);

    // ---- hoisted GEMMs: att_x and xk ----
    // att_x: 256x32 tiles = 8192 waves -> 1024 blocks of 8 waves
    gemm_bf16_packed<<<1024, 256, 0, stream>>>(xpack, p_attW, attb, attx,
                                               B_ * T_, U_, D_);
    // xk: 256x128 tiles = 32768 waves -> 4096 blocks
    gemm_bf16_packed<<<4096, 256, 0, stream>>>(xpack, p_kern, bias, xk,
                                               B_ * T_, G_, D_);

    zero_kernel<<<64, 256, 0, stream>>>(h, B_ * U_);
    zero_kernel<<<64, 256, 0, stream>>>(c, B_ * U_);

    // ---- sequential scan over T ----
    for (int t = 0; t < T_; ++t) {
        // hU = h @ attention_U : 2x32 = 64 tiles -> 8 blocks of 8 waves
        gemm_af32_bf16<<<8, 256, 0, stream>>>(h, p_attU, hU, B_, U_, U_);
        // scores: 4096 waves -> 512 blocks of 8 waves
        scores_kernel<<<512, 256, 0, stream>>>(attx, hU, attV, sc);
        // softmax + context
        softmax_z_kernel<<<B_, 128, 0, stream>>>(sc, x, z);
        // gates = xk_t + h@R + z@A : 2x128 = 256 tiles -> 32 blocks of 8 waves
        gates_gemm<<<32, 256, 0, stream>>>(h, z, p_R, p_A, xk, gates, t);
        // LSTM update + write h_t
        lstm_update<<<64, 256, 0, stream>>>(gates, h, c, out, t);
    }
}
